// GAT_45449343926639
// MI455X (gfx1250) — compile-verified, matrix operand order checked by reading, not run on hardware
//
#include <hip/hip_runtime.h>
#include <hip/hip_bf16.h>
#include <math.h>

#define N_NODES 50000
#define N_EDGES 800000
#define NEG_SLOPE 0.2f

typedef __attribute__((ext_vector_type(16))) _Float16 v16h;
typedef __attribute__((ext_vector_type(8)))  float    v8f;
typedef __attribute__((ext_vector_type(2)))  float    v2f;

union V16H { v16h v; _Float16 e[16]; };
union V8F  { v8f  v; float    e[8];  };
union V2F  { v2f  v; float    e[2];  };

#if __has_builtin(__builtin_amdgcn_wmma_f32_16x16x4_f32)
#define USE_WMMA_F32 1     // exact fp32 matrix path (V_WMMA_F32_16X16X4_F32)
#else
#define USE_WMMA_F32 0     // fallback: f16 inputs, f32 accumulate
#endif

// Monotone order-preserving f32 -> u32 encoding for atomicMax-based segment max.
__device__ __forceinline__ unsigned enc_f32(float f) {
  unsigned u = __float_as_uint(f);
  return (u & 0x80000000u) ? ~u : (u | 0x80000000u);
}
__device__ __forceinline__ float dec_f32(unsigned u) {
  u = (u & 0x80000000u) ? (u & 0x7FFFFFFFu) : ~u;
  return __uint_as_float(u);
}

// ---------------------------------------------------------------------------
// Init: zero the accumulators (harness poisons d_out/d_ws with 0xAA).
// ---------------------------------------------------------------------------
__global__ void gat_init(float* __restrict__ rst1, float* __restrict__ outp,
                         unsigned* __restrict__ m1u, float* __restrict__ s1,
                         unsigned* __restrict__ m2u, float* __restrict__ s2) {
  int i = blockIdx.x * blockDim.x + threadIdx.x;
  if (i < N_NODES * 128) { rst1[i] = 0.f; outp[i] = 0.f; }
  if (i < N_NODES * 4)   { m1u[i] = 0u; s1[i] = 0.f; m2u[i] = 0u; s2[i] = 0.f; }
}

// ---------------------------------------------------------------------------
// Generic 16x16 output tile GEMM helper: C = A(16xK) * B(K x 16),
// A row m at ap[m? ...]: lane supplies its own row pointer (already offset),
// B col n likewise (rows of W since B = W^T). Accumulates in f32.
//
// f32 path: V_WMMA_F32_16X16X4_F32, K-step 4.
//   A 16x4 (2 VGPRs): lane (m=lane&15, grp=lane>>4) -> A[m][2*grp + i]
//   B 4x16 (2 VGPRs): lane -> B[2*grp + i][n] = W[n][2*grp + i]
// f16 fallback: V_WMMA_F32_16X16X32_F16, K-step 32.
//   halves i=0..15 cover k = (i&7) + 8*grp + 16*(i>>3)
// ---------------------------------------------------------------------------
__device__ __forceinline__ v8f wmma_tile_rowrow(const float* __restrict__ hp,
                                                const float* __restrict__ wp,
                                                int grp, int K) {
  V8F acc = {};
#if USE_WMMA_F32
  #pragma unroll
  for (int k0 = 0; k0 < 128; k0 += 4) {
    if (k0 >= K) break;
    V2F a, b;
    int k = k0 + 2 * grp;
    a.e[0] = hp[k];     a.e[1] = hp[k + 1];
    b.e[0] = wp[k];     b.e[1] = wp[k + 1];
    acc.v = __builtin_amdgcn_wmma_f32_16x16x4_f32(false, a.v, false, b.v,
                                                  (short)0, acc.v, false, false);
  }
#else
  #pragma unroll
  for (int k0 = 0; k0 < 128; k0 += 32) {
    if (k0 >= K) break;
    V16H a, b;
    #pragma unroll
    for (int i = 0; i < 16; ++i) {
      int k = k0 + (i & 7) + 8 * grp + ((i >> 3) << 4);
      a.e[i] = (_Float16)hp[k];
      b.e[i] = (_Float16)wp[k];
    }
    acc.v = __builtin_amdgcn_wmma_f32_16x16x32_f16(false, a.v, false, b.v,
                                                   (short)0, acc.v, false, false);
  }
#endif
  return acc.v;
}

// ---------------------------------------------------------------------------
// Layer-1 GEMM: z1(N,128) = h(N,128) @ W1(128,128)^T. One wave per 16x16 tile.
// C/D layout: VGPR r, lane l -> element (M = r + 8*grp, N = lane&15).
// ---------------------------------------------------------------------------
__global__ void gat_gemm1_wmma(const float* __restrict__ h,
                               const float* __restrict__ W1,
                               float* __restrict__ z1) {
  const int wave = (blockIdx.x * blockDim.x + threadIdx.x) >> 5;
  const int lane = threadIdx.x & 31;
  if (wave >= (N_NODES / 16) * 8) return;           // uniform per-wave: EXEC stays all-1s
  const int rt  = wave >> 3;                        // row tile (3125)
  const int ct  = wave & 7;                         // col tile (8)
  const int grp = lane >> 4;
  const int mn  = lane & 15;
  const float* hp = h  + (rt * 16 + mn) * 128;
  const float* wp = W1 + (ct * 16 + mn) * 128;
  V8F acc; acc.v = wmma_tile_rowrow(hp, wp, grp, 128);
#pragma unroll
  for (int r = 0; r < 8; ++r)
    z1[(rt * 16 + r + 8 * grp) * 128 + ct * 16 + mn] = acc.e[r];
}

// ---------------------------------------------------------------------------
// Layer-2 GEMM: per head p, z2[:,p,:] = x[:,p,:] (N x 32) @ W2(32,32)^T. K=32.
// ---------------------------------------------------------------------------
__global__ void gat_gemm2_wmma(const float* __restrict__ x,
                               const float* __restrict__ W2,
                               float* __restrict__ z2) {
  const int wave = (blockIdx.x * blockDim.x + threadIdx.x) >> 5;
  const int lane = threadIdx.x & 31;
  if (wave >= (N_NODES / 16) * 4 * 2) return;
  const int rt   = wave >> 3;        // 3125 row tiles
  const int head = (wave >> 1) & 3;  // prefix dim (4)
  const int ct   = wave & 1;         // 2 col tiles of 16
  const int grp  = lane >> 4;
  const int mn   = lane & 15;
  const float* xp = x  + (rt * 16 + mn) * 128 + head * 32;
  const float* wp = W2 + (ct * 16 + mn) * 32;
  V8F acc; acc.v = wmma_tile_rowrow(xp, wp, grp, 32);
#pragma unroll
  for (int r = 0; r < 8; ++r)
    z2[(rt * 16 + r + 8 * grp) * 128 + head * 32 + ct * 16 + mn] = acc.e[r];
}

// ---------------------------------------------------------------------------
// Attention scores: el[n,h] = <z[n,h,:], al[h,:]>, er likewise.
// al_head_stride = 32 for layer 1 (al is (4,32)), 0 for layer 2 (al is (1,32)).
// ---------------------------------------------------------------------------
__global__ void gat_scores(const float* __restrict__ z,
                           const float* __restrict__ al,
                           const float* __restrict__ ar,
                           float* __restrict__ el, float* __restrict__ er,
                           int al_head_stride) {
  int idx = blockIdx.x * blockDim.x + threadIdx.x;   // n*4 + h
  if (idx >= N_NODES * 4) return;
  int hh = idx & 3;
  const float* zp  = z  + (idx >> 2) * 128 + hh * 32;
  const float* alp = al + hh * al_head_stride;
  const float* arp = ar + hh * al_head_stride;
  float sl = 0.f, sr = 0.f;
#pragma unroll
  for (int f = 0; f < 32; ++f) { sl += zp[f] * alp[f]; sr += zp[f] * arp[f]; }
  el[idx] = sl; er[idx] = sr;
}

// ---------------------------------------------------------------------------
// Edge pass A: segment max of leaky_relu(el[src]+er[dst]) over dst, per head.
// atomicMax on order-preserving uint encoding (no CAS loop).
// ---------------------------------------------------------------------------
__global__ void gat_edge_max(const int* __restrict__ src, const int* __restrict__ dst,
                             const float* __restrict__ el, const float* __restrict__ er,
                             unsigned* __restrict__ mu) {
  int gid = blockIdx.x * blockDim.x + threadIdx.x;
  if (gid >= N_EDGES * 4) return;
  int e = gid >> 2, hh = gid & 3;
  int s = src[e], d = dst[e];
  float v = el[s * 4 + hh] + er[d * 4 + hh];
  v = v > 0.f ? v : NEG_SLOPE * v;
  atomicMax(&mu[d * 4 + hh], enc_f32(v));
}

// ---------------------------------------------------------------------------
// Edge pass B (fused): ee = exp(lrelu(el+er) - m[dst]);
//   s[dst,h]     += ee                  (softmax denominator)
//   acc[dst,h,:] += ee * z[src,h,:]     (un-normalized aggregate)
// One wave32 per edge; lane = feature -> coalesced 128B gathers/scatters.
// ---------------------------------------------------------------------------
__global__ void gat_edge_acc(const int* __restrict__ src, const int* __restrict__ dst,
                             const float* __restrict__ el, const float* __restrict__ er,
                             const unsigned* __restrict__ mu,
                             const float* __restrict__ z,
                             float* __restrict__ ssum, float* __restrict__ acc) {
  int wave = (blockIdx.x * blockDim.x + threadIdx.x) >> 5;
  int lane = threadIdx.x & 31;
  if (wave >= N_EDGES) return;
  int s = src[wave], d = dst[wave];
  const float* zp = z + s * 128;
  float* ap = acc + d * 128;
#pragma unroll
  for (int hh = 0; hh < 4; ++hh) {
    float v = el[s * 4 + hh] + er[d * 4 + hh];           // uniform across lanes
    v = v > 0.f ? v : NEG_SLOPE * v;
    float ee = __expf(v - dec_f32(mu[d * 4 + hh]));
    if (lane == hh) atomicAdd(&ssum[d * 4 + hh], ee);    // lanes 0..3, one head each
    atomicAdd(&ap[hh * 32 + lane], ee * zp[hh * 32 + lane]);
  }
}

// ---------------------------------------------------------------------------
// Layer-1 epilogue: x = elu(acc / s + b1)   (s==0 only for isolated nodes)
// ---------------------------------------------------------------------------
__global__ void gat_norm_elu(const float* __restrict__ acc, const float* __restrict__ ssum,
                             const float* __restrict__ b1, float* __restrict__ x) {
  int i = blockIdx.x * blockDim.x + threadIdx.x;
  if (i >= N_NODES * 128) return;
  int j = i & 127;
  float sv = ssum[(i >> 7) * 4 + (j >> 5)];
  float v = (sv > 0.f ? acc[i] / sv : 0.f) + b1[j];
  x[i] = v > 0.f ? v : (__expf(v) - 1.f);
}

// ---------------------------------------------------------------------------
// Layer-2 epilogue (in place on d_out): out = out / s2 + b2  (b2 broadcast on f)
// ---------------------------------------------------------------------------
__global__ void gat_norm_out(const float* __restrict__ ssum, const float* __restrict__ b2,
                             float* __restrict__ outp) {
  int i = blockIdx.x * blockDim.x + threadIdx.x;
  if (i >= N_NODES * 128) return;
  int j = i & 127;
  float sv = ssum[(i >> 7) * 4 + (j >> 5)];
  outp[i] = (sv > 0.f ? outp[i] / sv : 0.f) + b2[j & 31];
}

// ---------------------------------------------------------------------------
extern "C" void kernel_launch(void* const* d_in, const int* in_sizes, int n_in,
                              void* d_out, int out_size, void* d_ws, size_t ws_size,
                              hipStream_t stream) {
  const float* h   = (const float*)d_in[0];
  const int*   src = (const int*)  d_in[1];
  const int*   dst = (const int*)  d_in[2];
  const float* W1  = (const float*)d_in[3];
  const float* al1 = (const float*)d_in[4];
  const float* ar1 = (const float*)d_in[5];
  const float* b1  = (const float*)d_in[6];
  const float* W2  = (const float*)d_in[7];
  const float* al2 = (const float*)d_in[8];
  const float* ar2 = (const float*)d_in[9];
  const float* b2  = (const float*)d_in[10];
  float* outp = (float*)d_out;

  // Workspace carve-up (all f32-sized slots).
  const size_t NB = (size_t)N_NODES * 128;   // big arrays
  const size_t NS = (size_t)N_NODES * 4;     // small per-(node,head) arrays
  float* ws  = (float*)d_ws;
  float* z1   = ws;            ws += NB;
  float* rst1 = ws;            ws += NB;
  float* x    = ws;            ws += NB;
  float* z2   = ws;            ws += NB;
  float* el1  = ws;            ws += NS;
  float* er1  = ws;            ws += NS;
  float* s1   = ws;            ws += NS;
  unsigned* m1u = (unsigned*)ws; ws += NS;
  float* el2  = ws;            ws += NS;
  float* er2  = ws;            ws += NS;
  float* s2   = ws;            ws += NS;
  unsigned* m2u = (unsigned*)ws; ws += NS;

  const int T = 256;
  const int big_blocks   = (N_NODES * 128 + T - 1) / T;      // 25000
  const int score_blocks = (N_NODES * 4 + T - 1) / T;
  const int emax_blocks  = (N_EDGES * 4 + T - 1) / T;
  const int eacc_blocks  = (N_EDGES * 32 + T - 1) / T;       // one wave32 per edge
  const int gemm_blocks  = ((N_NODES / 16) * 8 + 3) / 4;     // 4 waves / 128-thread block

  gat_init<<<big_blocks, T, 0, stream>>>(rst1, outp, m1u, s1, m2u, s2);

  // ---- layer 1 ----
  gat_gemm1_wmma<<<gemm_blocks, 128, 0, stream>>>(h, W1, z1);
  gat_scores<<<score_blocks, T, 0, stream>>>(z1, al1, ar1, el1, er1, 32);
  gat_edge_max<<<emax_blocks, T, 0, stream>>>(src, dst, el1, er1, m1u);
  gat_edge_acc<<<eacc_blocks, T, 0, stream>>>(src, dst, el1, er1, m1u, z1, s1, rst1);
  gat_norm_elu<<<big_blocks, T, 0, stream>>>(rst1, s1, b1, x);

  // ---- layer 2 ----
  gat_gemm2_wmma<<<gemm_blocks, 128, 0, stream>>>(x, W2, z2);
  gat_scores<<<score_blocks, T, 0, stream>>>(z2, al2, ar2, el2, er2, 0);
  gat_edge_max<<<emax_blocks, T, 0, stream>>>(src, dst, el2, er2, m2u);
  gat_edge_acc<<<eacc_blocks, T, 0, stream>>>(src, dst, el2, er2, m2u, z2, s2, outp);
  gat_norm_out<<<big_blocks, T, 0, stream>>>(s2, b2, outp);
}
